// TransformerTranslator_80118319940086
// MI455X (gfx1250) — matrix operand order repaired and checked by reading, not compiled, not run on hardware
//
#include <hip/hip_runtime.h>
#include <hip/hip_bf16.h>
#include <math.h>

typedef __attribute__((ext_vector_type(16))) _Float16 v16h;
typedef __attribute__((ext_vector_type(8)))  _Float16 v8h;
typedef __attribute__((ext_vector_type(8)))  float    v8f;

#define D_MODEL 1024
#define N_HEADS 16
#define DH      64
#define FF_DIM  4096
#define SEQ     512
#define BATCH   8
#define M_TOK   (BATCH*SEQ)   /* 4096 */
#define VOCAB   32000
#define NEGB    (-1e9f)

/* ------------------------------------------------------------------ */
/* Generic GEMM: Y[M,N] = act(X[M,K] @ W[N,K]^T + bias[N])            */
/* fp32 in/out, f16 WMMA fp32-acc.                                    */
/* 128x128 block tile, 8 waves as 4(M)x2(N); each wave owns a 32x64   */
/* C macro-tile (8 accumulators, 8 wmma / K-step).                    */
/* Double-buffered LDS: next K-slab global loads issued before the    */
/* current slab's WMMAs; converted + stored to LDS after compute.     */
/* REQUIRES: M % 128 == 0, N % 128 == 0, K % 32 == 0 (true for all    */
/* shapes in this model) -> fully branch-free staging & epilogue.     */
/* ------------------------------------------------------------------ */
#define BM 128
#define BN 128
#define BK 32
#define LDP (BK + 8)   /* LDS row stride in halves: 40 -> 80B (16B-mult) */

__device__ __forceinline__ v8h pack8(float4 u0, float4 u1) {
  v8h h;
  h[0] = (_Float16)u0.x; h[1] = (_Float16)u0.y;
  h[2] = (_Float16)u0.z; h[3] = (_Float16)u0.w;
  h[4] = (_Float16)u1.x; h[5] = (_Float16)u1.y;
  h[6] = (_Float16)u1.z; h[7] = (_Float16)u1.w;
  return h;
}

__global__ __launch_bounds__(256) void gemm_xwT(
    const float* __restrict__ X, int ldx,
    const float* __restrict__ W,     /* [N,K], row stride K */
    const float* __restrict__ bias,  /* [N] */
    float* __restrict__ Y, int ldy,
    int K, int relu)
{
  __shared__ _Float16 As[2][BM][LDP];
  __shared__ _Float16 Bs[2][BN][LDP];

  const int tid  = threadIdx.x;
  const int lane = tid & 31;
  const int wave = tid >> 5;
  const int waveM = wave & 3;           /* 0..3 -> 32-row strip   */
  const int waveN = wave >> 2;          /* 0..1 -> 64-col strip   */
  const int bm = blockIdx.y * BM;
  const int bn = blockIdx.x * BN;

  /* staging: each thread moves 16 A elems + 16 B elems per K-slab */
  const int srow = tid >> 1;            /* 0..127 */
  const int scol = (tid & 1) * 16;      /* 0 or 16 */

  const float* xp = X + (size_t)(bm + srow) * ldx + scol;
  const float* wp = W + (size_t)(bn + srow) * K + scol;

  v8f zero = {};
  v8f acc[2][4];
#pragma unroll
  for (int mt = 0; mt < 2; ++mt)
#pragma unroll
    for (int nt = 0; nt < 4; ++nt) acc[mt][nt] = zero;

  const int am0 = (waveM << 5) + (lane & 15);
  const int kb  = (lane & 16) ? 16 : 0;
  const int kbb = (lane & 16) ? 16 : 0;

  /* ---- prologue: stage slab 0 into buffer 0 ---- */
  {
    const float4* xv = (const float4*)xp;
    *(v8h*)&As[0][srow][scol]     = pack8(xv[0], xv[1]);
    *(v8h*)&As[0][srow][scol + 8] = pack8(xv[2], xv[3]);
    const float4* wv = (const float4*)wp;
    *(v8h*)&Bs[0][srow][scol]     = pack8(wv[0], wv[1]);
    *(v8h*)&Bs[0][srow][scol + 8] = pack8(wv[2], wv[3]);
  }
  __syncthreads();

  int buf = 0;
  for (int k0 = 0; k0 < K; k0 += BK) {
    const bool more = (k0 + BK) < K;
    /* issue next slab's global loads BEFORE this slab's compute */
    float4 na0, na1, na2, na3, nb0, nb1, nb2, nb3;
    if (more) {
      const float4* xv = (const float4*)(xp + k0 + BK);
      na0 = xv[0]; na1 = xv[1]; na2 = xv[2]; na3 = xv[3];
      const float4* wv = (const float4*)(wp + k0 + BK);
      nb0 = wv[0]; nb1 = wv[1]; nb2 = wv[2]; nb3 = wv[3];
      if (k0 + 2 * BK < K) {            /* global_prefetch_b8 */
        __builtin_prefetch(xp + k0 + 2 * BK);
        __builtin_prefetch(wp + k0 + 2 * BK);
      }
    }

    /* ---- fragments + 8 WMMAs from current buffer ---- */
    v16h a[2], bf[4];
#pragma unroll
    for (int mt = 0; mt < 2; ++mt) {
      const int r = am0 + (mt << 4);
#pragma unroll
      for (int j = 0; j < 8; ++j) {
        a[mt][j]     = As[buf][r][kb + j];
        a[mt][j + 8] = As[buf][r][kb + 8 + j];
      }
    }
#pragma unroll
    for (int nt = 0; nt < 4; ++nt) {
      const int c = (waveN << 6) + (nt << 4) + (lane & 15);
#pragma unroll
      for (int j = 0; j < 16; ++j) bf[nt][j] = Bs[buf][c][kbb + j];
    }
#pragma unroll
    for (int mt = 0; mt < 2; ++mt)
#pragma unroll
      for (int nt = 0; nt < 4; ++nt)
        acc[mt][nt] = __builtin_amdgcn_wmma_f32_16x16x32_f16(
            false, a[mt], false, bf[nt], (short)0, acc[mt][nt], false, false);

    /* ---- convert + store next slab into the other buffer ---- */
    if (more) {
      const int nb = buf ^ 1;
      *(v8h*)&As[nb][srow][scol]     = pack8(na0, na1);
      *(v8h*)&As[nb][srow][scol + 8] = pack8(na2, na3);
      *(v8h*)&Bs[nb][srow][scol]     = pack8(nb0, nb1);
      *(v8h*)&Bs[nb][srow][scol + 8] = pack8(nb2, nb3);
    }
    __syncthreads();
    buf ^= 1;
  }

  /* ---- epilogue (no guards: tile-divisible shapes) ---- */
#pragma unroll
  for (int nt = 0; nt < 4; ++nt) {
    const int gcol = bn + (waveN << 6) + (nt << 4) + (lane & 15);
    const float bv = bias[gcol];
#pragma unroll
    for (int mt = 0; mt < 2; ++mt) {
#pragma unroll
      for (int r = 0; r < 8; ++r) {
        const int grow = bm + (waveM << 5) + (mt << 4) + r + ((lane & 16) ? 8 : 0);
        float v = acc[mt][nt][r] + bv;
        if (relu) v = fmaxf(v, 0.0f);
        Y[(size_t)grow * ldy + gcol] = v;
      }
    }
  }
}

/* ------------------------------------------------------------------ */
/* scores[b,h,q,k] = (Q.K^T)/8 + pad/causal bias.  Q/K in fused qkv.  */
/* ------------------------------------------------------------------ */
__global__ __launch_bounds__(256) void attn_scores(
    const float* __restrict__ qkv, const int* __restrict__ key_ids,
    float* __restrict__ scores, int causal)
{
  const int lane = threadIdx.x & 31, wave = threadIdx.x >> 5;
  const int ktile = blockIdx.x;
  const int qtile = blockIdx.y * 8 + wave;
  const int bh = blockIdx.z, b = bh >> 4, h = bh & 15;
  const int kb  = (lane & 16) ? 16 : 0;
  const int kbb = (lane & 16) ? 16 : 0;
  const int qrow = (qtile << 4) + (lane & 15);
  const int kpos = (ktile << 4) + (lane & 15);
  const size_t qbase = (size_t)(b * SEQ + qrow) * (3 * D_MODEL) + h * DH;
  const size_t kbase = (size_t)(b * SEQ + kpos) * (3 * D_MODEL) + D_MODEL + h * DH;

  v8f acc = {};
#pragma unroll
  for (int k0 = 0; k0 < DH; k0 += 32) {
    v16h a, bb;
#pragma unroll
    for (int j = 0; j < 8; ++j) {
      a[j]     = (_Float16)qkv[qbase + k0 + kb + j];
      a[j + 8] = (_Float16)qkv[qbase + k0 + kb + 8 + j];
    }
#pragma unroll
    for (int j = 0; j < 16; ++j)
      bb[j] = (_Float16)qkv[kbase + k0 + kbb + j];
    acc = __builtin_amdgcn_wmma_f32_16x16x32_f16(
        false, a, false, bb, (short)0, acc, false, false);
  }
  const float padb = (key_ids[b * SEQ + kpos] == 0) ? NEGB : 0.0f;
#pragma unroll
  for (int r = 0; r < 8; ++r) {
    const int qpos = (qtile << 4) + r + ((lane & 16) ? 8 : 0);
    float v = acc[r] * 0.125f + padb;
    if (causal && kpos > qpos) v += NEGB;
    scores[((size_t)bh * SEQ + qpos) * SEQ + kpos] = v;
  }
}

/* one block per score row of length 512 */
__global__ __launch_bounds__(256) void softmax_rows(float* __restrict__ s)
{
  __shared__ float red[256];
  float* row = s + (size_t)blockIdx.x * SEQ;
  const int t = threadIdx.x;
  float a = row[t], b = row[t + 256];
  red[t] = fmaxf(a, b); __syncthreads();
  for (int off = 128; off > 0; off >>= 1) {
    if (t < off) red[t] = fmaxf(red[t], red[t + off]);
    __syncthreads();
  }
  const float m = red[0]; __syncthreads();
  const float e0 = __expf(a - m), e1 = __expf(b - m);
  red[t] = e0 + e1; __syncthreads();
  for (int off = 128; off > 0; off >>= 1) {
    if (t < off) red[t] += red[t + off];
    __syncthreads();
  }
  const float inv = 1.0f / red[0];
  row[t] = e0 * inv; row[t + 256] = e1 * inv;
}

/* ctx[b,q,h*64+dd] = sum_k att[b,h,q,k] * V[b,k,h,dd]                */
__global__ __launch_bounds__(256) void attn_av(
    const float* __restrict__ att, const float* __restrict__ qkv,
    float* __restrict__ ctx)
{
  const int lane = threadIdx.x & 31, wave = threadIdx.x >> 5;
  const int ntile = wave & 3;
  const int mtile = blockIdx.x * 2 + (wave >> 2);
  const int bh = blockIdx.y, b = bh >> 4, h = bh & 15;
  const int kb  = (lane & 16) ? 16 : 0;
  const int kbb = (lane & 16) ? 16 : 0;
  const int qrow = (mtile << 4) + (lane & 15);
  const int dd   = (ntile << 4) + (lane & 15);
  const float* arow = att + ((size_t)bh * SEQ + qrow) * SEQ;

  v8f acc = {};
  for (int k0 = 0; k0 < SEQ; k0 += 32) {
    v16h a, bb;
#pragma unroll
    for (int j = 0; j < 8; ++j) {
      a[j]     = (_Float16)arow[k0 + kb + j];
      a[j + 8] = (_Float16)arow[k0 + kb + 8 + j];
    }
#pragma unroll
    for (int j = 0; j < 16; ++j)
      bb[j] = (_Float16)qkv[(size_t)(b * SEQ + k0 + kbb + j) * (3 * D_MODEL)
                            + 2 * D_MODEL + h * DH + dd];
    acc = __builtin_amdgcn_wmma_f32_16x16x32_f16(
        false, a, false, bb, (short)0, acc, false, false);
  }
#pragma unroll
  for (int r = 0; r < 8; ++r) {
    const int qpos = (mtile << 4) + r + ((lane & 16) ? 8 : 0);
    ctx[(size_t)(b * SEQ + qpos) * D_MODEL + h * DH + dd] = acc[r];
  }
}

/* x = LN(x + sub) * g + b   (post-norm residual), one block per row  */
__global__ __launch_bounds__(256) void add_layernorm(
    float* __restrict__ x, const float* __restrict__ sub,
    const float* __restrict__ g, const float* __restrict__ b)
{
  __shared__ float red[256];
  const size_t base = (size_t)blockIdx.x * D_MODEL;
  const int t = threadIdx.x;
  float v[4]; float s = 0.0f;
#pragma unroll
  for (int i = 0; i < 4; ++i) {
    const int c = t + i * 256;
    v[i] = x[base + c] + sub[base + c];
    s += v[i];
  }
  red[t] = s; __syncthreads();
  for (int off = 128; off > 0; off >>= 1) {
    if (t < off) red[t] += red[t + off];
    __syncthreads();
  }
  const float mean = red[0] * (1.0f / D_MODEL); __syncthreads();
  s = 0.0f;
#pragma unroll
  for (int i = 0; i < 4; ++i) { const float d = v[i] - mean; s += d * d; }
  red[t] = s; __syncthreads();
  for (int off = 128; off > 0; off >>= 1) {
    if (t < off) red[t] += red[t + off];
    __syncthreads();
  }
  const float rstd = rsqrtf(red[0] * (1.0f / D_MODEL) + 1e-5f);
#pragma unroll
  for (int i = 0; i < 4; ++i) {
    const int c = t + i * 256;
    x[base + c] = (v[i] - mean) * rstd * g[c] + b[c];
  }
}

/* x[row] = emb[ids[row]]*sqrt(d) + sinusoidal PE                     */
__global__ __launch_bounds__(256) void embed_pos(
    const int* __restrict__ ids, const float* __restrict__ emb,
    float* __restrict__ x)
{
  const int row = blockIdx.x;
  const int spos = row & (SEQ - 1);
  const size_t base = (size_t)row * D_MODEL;
  const float* erow = emb + (size_t)ids[row] * D_MODEL;
#pragma unroll
  for (int i = 0; i < 4; ++i) {
    const int c = threadIdx.x + i * 256;
    const int i2 = c & ~1;
    const float angle = (float)spos * __powf(10000.0f, -(float)i2 / (float)D_MODEL);
    const float pe = (c & 1) ? __cosf(angle) : __sinf(angle);
    x[base + c] = erow[c] * 32.0f + pe;   /* sqrt(1024) = 32 */
  }
}

/* ------------------------------------------------------------------ */
extern "C" void kernel_launch(void* const* d_in, const int* in_sizes, int n_in,
                              void* d_out, int out_size, void* d_ws, size_t ws_size,
                              hipStream_t stream) {
  (void)in_sizes; (void)n_in; (void)out_size; (void)ws_size;
  const int* inp_ids = (const int*)d_in[0];
  const int* tgt_ids = (const int*)d_in[1];
  /* enc_params, dict insertion order */
  const float* eWqkv = (const float*)d_in[2];
  const float* ebqkv = (const float*)d_in[3];
  const float* eWo   = (const float*)d_in[4];
  const float* ebo   = (const float*)d_in[5];
  const float* eW1   = (const float*)d_in[6];
  const float* eb1   = (const float*)d_in[7];
  const float* eW2   = (const float*)d_in[8];
  const float* eb2   = (const float*)d_in[9];
  const float* eln1g = (const float*)d_in[10];
  const float* eln1b = (const float*)d_in[11];
  const float* eln2g = (const float*)d_in[12];
  const float* eln2b = (const float*)d_in[13];
  /* dec_params */
  const float* dWsaqkv = (const float*)d_in[14];
  const float* dbsaqkv = (const float*)d_in[15];
  const float* dWsao   = (const float*)d_in[16];
  const float* dbsao   = (const float*)d_in[17];
  const float* dWcaqkv = (const float*)d_in[18];
  const float* dbcaqkv = (const float*)d_in[19];
  const float* dWcao   = (const float*)d_in[20];
  const float* dbcao   = (const float*)d_in[21];
  const float* dW1     = (const float*)d_in[22];
  const float* db1     = (const float*)d_in[23];
  const float* dW2     = (const float*)d_in[24];
  const float* db2     = (const float*)d_in[25];
  const float* dln1g   = (const float*)d_in[26];
  const float* dln1b   = (const float*)d_in[27];
  const float* dln2g   = (const float*)d_in[28];
  const float* dln2b   = (const float*)d_in[29];
  const float* dln3g   = (const float*)d_in[30];
  const float* dln3b   = (const float*)d_in[31];
  const float* enc_emb = (const float*)d_in[32];
  const float* dec_emb = (const float*)d_in[33];
  const float* out_W   = (const float*)d_in[34];
  const float* out_b   = (const float*)d_in[35];

  float* ws = (float*)d_ws;
  float* x_enc  = ws; ws += (size_t)M_TOK * D_MODEL;
  float* x_dec  = ws; ws += (size_t)M_TOK * D_MODEL;
  float* qkv    = ws; ws += (size_t)M_TOK * 3 * D_MODEL;
  float* scores = ws; ws += (size_t)BATCH * N_HEADS * SEQ * SEQ;
  float* ctx    = ws; ws += (size_t)M_TOK * D_MODEL;
  float* ffh    = ws; ws += (size_t)M_TOK * FF_DIM;
  float* tmp    = ws; ws += (size_t)M_TOK * D_MODEL;

  const size_t sW3 = (size_t)3 * D_MODEL * D_MODEL, sb3 = 3 * D_MODEL;
  const size_t sWd = (size_t)D_MODEL * D_MODEL,     sbd = D_MODEL;
  const size_t sWf = (size_t)FF_DIM * D_MODEL,      sbf = FF_DIM;

  dim3 blk(256);
  auto gemm = [&](const float* X, int ldx, const float* W, const float* bias,
                  float* Y, int ldy, int M, int N, int K, int relu) {
    dim3 g(N / BN, M / BM);
    gemm_xwT<<<g, blk, 0, stream>>>(X, ldx, W, bias, Y, ldy, K, relu);
  };
  auto attn = [&](const int* kids, int causal, float* ctxp) {
    attn_scores<<<dim3(SEQ / 16, SEQ / 128, BATCH * N_HEADS), blk, 0, stream>>>(
        qkv, kids, scores, causal);
    softmax_rows<<<dim3(BATCH * N_HEADS * SEQ), blk, 0, stream>>>(scores);
    attn_av<<<dim3(SEQ / 32, BATCH * N_HEADS), blk, 0, stream>>>(scores, qkv, ctxp);
  };

  /* ---------------- encoder ---------------- */
  embed_pos<<<dim3(M_TOK), blk, 0, stream>>>(inp_ids, enc_emb, x_enc);
  for (int l = 0; l < 6; ++l) {
    gemm(x_enc, D_MODEL, eWqkv + l * sW3, ebqkv + l * sb3,
         qkv, 3 * D_MODEL, M_TOK, 3 * D_MODEL, D_MODEL, 0);
    attn(inp_ids, 0, ctx);
    gemm(ctx, D_MODEL, eWo + l * sWd, ebo + l * sbd,
         tmp, D_MODEL, M_TOK, D_MODEL, D_MODEL, 0);
    add_layernorm<<<dim3(M_TOK), blk, 0, stream>>>(x_enc, tmp, eln1g + l * sbd, eln1b + l * sbd);
    gemm(x_enc, D_MODEL, eW1 + l * sWf, eb1 + l * sbf,
         ffh, FF_DIM, M_TOK, FF_DIM, D_MODEL, 1);
    gemm(ffh, FF_DIM, eW2 + l * sWf, eb2 + l * sbd,
         tmp, D_MODEL, M_TOK, D_MODEL, FF_DIM, 0);
    add_layernorm<<<dim3(M_TOK), blk, 0, stream>>>(x_enc, tmp, eln2g + l * sbd, eln2b + l * sbd);
  }

  /* ---------------- decoder ---------------- */
  embed_pos<<<dim3(M_TOK), blk, 0, stream>>>(tgt_ids, dec_emb, x_dec);
  for (int l = 0; l < 6; ++l) {
    /* masked self-attention */
    gemm(x_dec, D_MODEL, dWsaqkv + l * sW3, dbsaqkv + l * sb3,
         qkv, 3 * D_MODEL, M_TOK, 3 * D_MODEL, D_MODEL, 0);
    attn(tgt_ids, 1, ctx);
    gemm(ctx, D_MODEL, dWsao + l * sWd, dbsao + l * sbd,
         tmp, D_MODEL, M_TOK, D_MODEL, D_MODEL, 0);
    add_layernorm<<<dim3(M_TOK), blk, 0, stream>>>(x_dec, tmp, dln1g + l * sbd, dln1b + l * sbd);
    /* cross-attention: Q from x_dec, K/V from encoder memory */
    gemm(x_dec, D_MODEL, dWcaqkv + l * sW3, dbcaqkv + l * sb3,
         qkv, 3 * D_MODEL, M_TOK, D_MODEL, D_MODEL, 0);
    gemm(x_enc, D_MODEL, dWcaqkv + l * sW3 + (size_t)D_MODEL * D_MODEL,
         dbcaqkv + l * sb3 + D_MODEL,
         qkv + D_MODEL, 3 * D_MODEL, M_TOK, 2 * D_MODEL, D_MODEL, 0);
    attn(inp_ids, 0, ctx);
    gemm(ctx, D_MODEL, dWcao + l * sWd, dbcao + l * sbd,
         tmp, D_MODEL, M_TOK, D_MODEL, D_MODEL, 0);
    add_layernorm<<<dim3(M_TOK), blk, 0, stream>>>(x_dec, tmp, dln2g + l * sbd, dln2b + l * sbd);
    /* FFN */
    gemm(x_dec, D_MODEL, dW1 + l * sWf, db1 + l * sbf,
         ffh, FF_DIM, M_TOK, FF_DIM, D_MODEL, 1);
    gemm(ffh, FF_DIM, dW2 + l * sWf, db2 + l * sbd,
         tmp, D_MODEL, M_TOK, D_MODEL, FF_DIM, 0);
    add_layernorm<<<dim3(M_TOK), blk, 0, stream>>>(x_dec, tmp, dln3g + l * sbd, dln3b + l * sbd);
  }

  /* logits = x_dec @ out_W^T + out_b   (4096 x 32000 x 1024) */
  gemm(x_dec, D_MODEL, out_W, out_b, (float*)d_out, VOCAB, M_TOK, VOCAB, D_MODEL, 0);
}
